// ALiBiAttention_29532195127744
// MI455X (gfx1250) — matrix operand order repaired and checked by reading, not compiled, not run on hardware
//
#include <hip/hip_runtime.h>
#include <stdint.h>

// ---------------------------------------------------------------------------
// ALiBi causal attention for MI455X (gfx1250), bf16 WMMA pipeline.
// B=4, T=2048, C=672, H=14, D=48.
// Double-buffered async (ASYNCcnt) global->LDS staging overlapped with WMMA.
// ---------------------------------------------------------------------------

typedef __attribute__((ext_vector_type(16))) __bf16 v16bf;
typedef __attribute__((ext_vector_type(8)))  float  v8f;
typedef int vint4 __attribute__((vector_size(16)));   // matches builtin param type

union FragB { v16bf v; uint32_t u[8]; uint4 q[2]; };
union AccF  { v8f v; float f[8]; };

// --- CDNA5 async global->LDS path (guarded; sync fallback) -----------------
#if defined(__HIP_DEVICE_COMPILE__) && __has_builtin(__builtin_amdgcn_global_load_async_to_lds_b128)
#define HAS_ASYNC_B128 1
#endif
#if defined(__HIP_DEVICE_COMPILE__) && __has_builtin(__builtin_amdgcn_global_load_async_to_lds_b32)
#define HAS_ASYNC_B32 1
#endif

#define GPTR128(p) ((__attribute__((address_space(1))) vint4*)(p))
#define LPTR128(p) ((__attribute__((address_space(3))) vint4*)(p))
#define GPTR32(p)  ((__attribute__((address_space(1))) int*)(p))
#define LPTR32(p)  ((__attribute__((address_space(3))) int*)(p))

__device__ __forceinline__ void stage_b128(const unsigned short* g, unsigned short* l) {
#if defined(HAS_ASYNC_B128)
  __builtin_amdgcn_global_load_async_to_lds_b128(GPTR128(g), LPTR128(l), 0, 0);
#else
  *reinterpret_cast<uint4*>(l) = *reinterpret_cast<const uint4*>(g);
#endif
}

__device__ __forceinline__ void stage_b32(const unsigned short* g, unsigned short* l) {
#if defined(HAS_ASYNC_B32)
  __builtin_amdgcn_global_load_async_to_lds_b32(GPTR32(g), LPTR32(l), 0, 0);
#else
  *reinterpret_cast<uint32_t*>(l) = *reinterpret_cast<const uint32_t*>(g);
#endif
}

__device__ __forceinline__ void async_wait_all() {
#if defined(__HIP_DEVICE_COMPILE__)
#if __has_builtin(__builtin_amdgcn_s_wait_asynccnt)
  __builtin_amdgcn_s_wait_asynccnt(0);
#else
  asm volatile("s_wait_asynccnt 0x0" ::: "memory");
#endif
#endif
}

__device__ __forceinline__ unsigned short f2bf(float x) {
  uint32_t u = __builtin_bit_cast(uint32_t, x);
  u = (u + 0x7FFFu + ((u >> 16) & 1u)) >> 16;           // round-to-nearest-even
  return (unsigned short)u;
}
__device__ __forceinline__ float bf2f(unsigned short h) {
  uint32_t u = ((uint32_t)h) << 16;
  return __builtin_bit_cast(float, u);
}

// A-matrix 16x32 bf16 fragment K position: lanes 0-15 hold
// K={0..15}, lanes 16-31 hold K={16..31}, 2 elems per VGPR.
__device__ __forceinline__ int kposf(int j, int hf) {
  return (2 * j + 16 * hf);
}

// Load 16x32 A-fragment from LDS tile stored row-major [row][k] (stride even).
__device__ __forceinline__ v16bf ldsFragA(const unsigned short* tile, int stride) {
  int lane = threadIdx.x & 31;
  const unsigned short* p = tile + (lane & 15) * stride;
  int hf = lane >> 4;
  FragB f;
#pragma unroll
  for (int j = 0; j < 8; ++j)
    f.u[j] = *reinterpret_cast<const uint32_t*>(p + kposf(j, hf));
  return f.v;
}

// Load 32x16 B-fragment from LDS tile stored [n][k] row-major: lane holds
// column n = lane&15, contiguous K range 16*(lane>>4)..+15.
__device__ __forceinline__ v16bf ldsFragB(const unsigned short* tile, int stride) {
  int lane = threadIdx.x & 31;
  const unsigned short* p = tile + (lane & 15) * stride + 16 * (lane >> 4);
  FragB f;
  f.q[0] = *reinterpret_cast<const uint4*>(p);
  f.q[1] = *reinterpret_cast<const uint4*>(p + 8);
  return f.v;
}

// B-fragment with zero-fill for K-dim positions >= dMax (D=48 padded to 64).
__device__ __forceinline__ v16bf ldsFragBZ(const unsigned short* tile, int stride,
                                           int colOff, int dMax) {
  int lane = threadIdx.x & 31;
  int base = colOff + 16 * (lane >> 4);
  FragB f;
  if (base < dMax) {
    const unsigned short* p = tile + (lane & 15) * stride + base;
    f.q[0] = *reinterpret_cast<const uint4*>(p);
    f.q[1] = *reinterpret_cast<const uint4*>(p + 8);
  } else {
    f.q[0] = make_uint4(0u, 0u, 0u, 0u);
    f.q[1] = make_uint4(0u, 0u, 0u, 0u);
  }
  return f.v;
}

// ---------------------------------------------------------------------------
// Converters
// ---------------------------------------------------------------------------
__global__ void cvt_f32_bf16(const float* __restrict__ in,
                             unsigned short* __restrict__ out, int n) {
  for (int i = blockIdx.x * blockDim.x + threadIdx.x; i < n;
       i += gridDim.x * blockDim.x)
    out[i] = f2bf(in[i]);
}

// in[rows][cols] f32 -> out[cols][rows] bf16 (N-major weights)
__global__ void cvt_transpose_bf16(const float* __restrict__ in,
                                   unsigned short* __restrict__ out,
                                   int rows, int cols) {
  int n = rows * cols;
  for (int i = blockIdx.x * blockDim.x + threadIdx.x; i < n;
       i += gridDim.x * blockDim.x) {
    int r = i / cols, c = i % cols;
    out[(size_t)c * rows + r] = f2bf(in[i]);
  }
}

// ---------------------------------------------------------------------------
// GEMM: C[M,N] (bf16) = A[M,K] (bf16, row-major) x Bt[N,K]^T (bf16, N-major)
// Block: 256 threads (8 waves). Block tile 512(M) x 32(N), K-step 32,
// double-buffered LDS: async stage of step s+1 overlaps WMMA of step s.
// Requires M%512==0, N%32==0, K%32==0. (8192 / 2016|672 / 672 all satisfy.)
// ---------------------------------------------------------------------------
constexpr int AST = 40;  // 80B rows: 16B-aligned, 20-dword bank stride
constexpr int BST = 40;
constexpr int ATILE = 512 * AST;
constexpr int BTILE = 32 * BST;

__device__ __forceinline__ void gemm_stage(const unsigned short* __restrict__ A,
                                           const unsigned short* __restrict__ Bt,
                                           unsigned short* As, unsigned short* Bs,
                                           int m0, int n0, int ks, int K, int tid) {
#pragma unroll
  for (int i = 0; i < 8; ++i) {
    int c = tid + i * 256;
    int row = c >> 2, c8 = (c & 3) * 8;
    stage_b128(A + (size_t)(m0 + row) * K + ks + c8, As + row * AST + c8);
  }
  if (tid < 128) {
    int row = tid >> 2, c8 = (tid & 3) * 8;
    stage_b128(Bt + (size_t)(n0 + row) * K + ks + c8, Bs + row * BST + c8);
  }
}

__global__ __launch_bounds__(256) void gemm_bf16_wmma(
    const unsigned short* __restrict__ A,
    const unsigned short* __restrict__ Bt,
    unsigned short* __restrict__ C, int K, int N) {
  __shared__ __align__(16) unsigned short As[2 * ATILE];   // 80 KB
  __shared__ __align__(16) unsigned short Bs[2 * BTILE];   // 5 KB

  const int tid = threadIdx.x;
  const int wave = tid >> 5, lane = tid & 31;
  const int m0 = blockIdx.y * 512;
  const int n0 = blockIdx.x * 32;

  AccF acc[4][2];
#pragma unroll
  for (int a = 0; a < 4; ++a)
#pragma unroll
    for (int b = 0; b < 2; ++b)
#pragma unroll
      for (int e = 0; e < 8; ++e) acc[a][b].f[e] = 0.0f;

  const int nsteps = K / 32;
  gemm_stage(A, Bt, As, Bs, m0, n0, 0, K, tid);            // prologue: stage step 0

  for (int s = 0; s < nsteps; ++s) {
    async_wait_all();                                      // stage(s) landed
    __syncthreads();                                       // visible to all waves
    unsigned short* Acur = As + (s & 1) * ATILE;
    unsigned short* Bcur = Bs + (s & 1) * BTILE;
    if (s + 1 < nsteps)                                    // overlap: stage(s+1)
      gemm_stage(A, Bt, As + ((s + 1) & 1) * ATILE, Bs + ((s + 1) & 1) * BTILE,
                 m0, n0, (s + 1) * 32, K, tid);

    v16bf b0 = ldsFragB(Bcur + 0 * BST, BST);
    v16bf b1 = ldsFragB(Bcur + 16 * BST, BST);
#pragma unroll
    for (int mi = 0; mi < 4; ++mi) {
      v16bf a = ldsFragA(Acur + (wave * 64 + mi * 16) * AST, AST);
      acc[mi][0].v = __builtin_amdgcn_wmma_f32_16x16x32_bf16(false, a, false, b0, (short)0, acc[mi][0].v, false, false);
      acc[mi][1].v = __builtin_amdgcn_wmma_f32_16x16x32_bf16(false, a, false, b1, (short)0, acc[mi][1].v, false, false);
    }
  }

  const int laneN = lane & 15, hf = lane >> 4;
#pragma unroll
  for (int mi = 0; mi < 4; ++mi)
#pragma unroll
    for (int ni = 0; ni < 2; ++ni)
#pragma unroll
      for (int v = 0; v < 8; ++v) {
        int m = m0 + wave * 64 + mi * 16 + hf * 8 + v;     // C/D layout (ISA 7.12.2)
        int n = n0 + ni * 16 + laneN;
        C[(size_t)m * N + n] = f2bf(acc[mi][ni].f[v]);
      }
}

// ---------------------------------------------------------------------------
// Flash-style causal ALiBi attention, one block per (b, h, 128 query rows).
// qkv: [B*T][2016] bf16  (q|k|v).  y: [B*T][672] bf16.
// K staged async (double-buffered); V transpose split around the WMMA core so
// its global-load latency is covered by compute.
// ---------------------------------------------------------------------------
__global__ __launch_bounds__(256) void attn_wmma(
    const unsigned short* __restrict__ qkv,
    unsigned short* __restrict__ y) {
  constexpr int KST = 56;  // 112B rows: 16B-aligned, conflict-free 28-dword stride
  constexpr int VST = 40;  // 80B rows
  constexpr int PST = 34;  // 17-dword stride, conflict-free for A-frag reads
  constexpr int KTILE = 32 * KST;
  constexpr int VTILE = 48 * VST;
  __shared__ __align__(16) unsigned short Ks[2 * KTILE];      // K tiles [key][d]
  __shared__ __align__(16) unsigned short Vt[2 * VTILE];      // V tiles transposed [d][key]
  __shared__ __align__(16) unsigned short Ps[8 * 16 * PST];   // per-wave P scratch

  const int tid = threadIdx.x;
  const int wave = tid >> 5, lane = tid & 31;
  const int r = lane & 15, hf = lane >> 4;
  const int b = blockIdx.z, h = blockIdx.y;
  const int qBase = blockIdx.x * 128;
  const int q0 = qBase + wave * 16;
  const size_t LD = 2016;

  const float scale = bf2f(f2bf(0.14433756730f));                        // 1/sqrt(48)
  const float slope = bf2f(f2bf(exp2f(-8.0f * (float)(h + 1) / 14.0f))); // ALiBi slope

  // Persistent Q A-fragments (16x64, zero-padded beyond D=48), loaded from global.
  FragB qa0, qa1;
  {
    const unsigned short* qrow = qkv + (size_t)(b * 2048 + q0 + r) * LD + h * 48;
#pragma unroll
    for (int j = 0; j < 8; ++j) {
      int k = kposf(j, hf);
      qa0.u[j] = *reinterpret_cast<const uint32_t*>(qrow + k);
      int k1 = 32 + k;
      qa1.u[j] = (k1 < 48) ? *reinterpret_cast<const uint32_t*>(qrow + k1) : 0u;
    }
  }

  AccF o[3];
  float mrow[8], lrow[8];
#pragma unroll
  for (int v = 0; v < 8; ++v) { mrow[v] = -3.0e38f; lrow[v] = 0.0f; }
#pragma unroll
  for (int nj = 0; nj < 3; ++nj)
#pragma unroll
    for (int v = 0; v < 8; ++v) o[nj].f[v] = 0.0f;

  const int kEnd = qBase + 128;                 // causal horizon for this block

  // Prologue: stage chunk 0 into buffer 0.
#pragma unroll
  for (int i = 0; i < 3; ++i) {
    int dw = tid + i * 256;                     // 768 dwords each for K and V
    int key = dw / 24, dp = dw % 24;
    size_t row = (size_t)(b * 2048 + key) * LD;
    stage_b32(qkv + row + 672 + h * 48 + dp * 2, Ks + key * KST + dp * 2);
    uint32_t vd = *reinterpret_cast<const uint32_t*>(qkv + row + 1344 + h * 48 + dp * 2);
    Vt[(dp * 2 + 0) * VST + key] = (unsigned short)(vd & 0xFFFFu);
    Vt[(dp * 2 + 1) * VST + key] = (unsigned short)(vd >> 16);
  }

  for (int kc = 0, st = 0; kc < kEnd; kc += 32, ++st) {
    async_wait_all();
    __syncthreads();
    const unsigned short* Kcur = Ks + (st & 1) * KTILE;
    const unsigned short* Vcur = Vt + (st & 1) * VTILE;
    unsigned short* Knxt = Ks + ((st + 1) & 1) * KTILE;
    unsigned short* Vnxt = Vt + ((st + 1) & 1) * VTILE;
    const bool haveNext = (kc + 32) < kEnd;

    // Issue next chunk's K async DMA + V global loads before compute.
    uint32_t vb[3];
    if (haveNext) {
#pragma unroll
      for (int i = 0; i < 3; ++i) {
        int dw = tid + i * 256;
        int key = dw / 24, dp = dw % 24;
        size_t row = (size_t)(b * 2048 + kc + 32 + key) * LD;
        stage_b32(qkv + row + 672 + h * 48 + dp * 2, Knxt + key * KST + dp * 2);
        vb[i] = *reinterpret_cast<const uint32_t*>(qkv + row + 1344 + h * 48 + dp * 2);
      }
    }

    if (kc <= q0 + 15) {          // skip compute (not staging) when fully masked
      // S = Q x K^T over two 16-key subtiles (K-dim 48 -> two k=32 WMMA).
      AccF s[2];
#pragma unroll
      for (int kt = 0; kt < 2; ++kt) {
        v16bf kb0 = ldsFragBZ(Kcur + kt * 16 * KST, KST, 0, 48);
        v16bf kb1 = ldsFragBZ(Kcur + kt * 16 * KST, KST, 32, 48);
        AccF z;
#pragma unroll
        for (int v = 0; v < 8; ++v) z.f[v] = 0.0f;
        z.v = __builtin_amdgcn_wmma_f32_16x16x32_bf16(false, qa1.v, false, kb1, (short)0, z.v, false, false);
        z.v = __builtin_amdgcn_wmma_f32_16x16x32_bf16(false, qa0.v, false, kb0, (short)0, z.v, false, false);
        s[kt] = z;
      }

      // scale + ALiBi bias + causal mask (element (m,n): m=hf*8+v, n=r).
#pragma unroll
      for (int kt = 0; kt < 2; ++kt)
#pragma unroll
        for (int v = 0; v < 8; ++v) {
          int qRow = q0 + hf * 8 + v;
          int key = kc + kt * 16 + r;
          float sv = s[kt].f[v] * scale;
          sv += fmaxf((float)(key - qRow), 0.0f) * slope;
          if (key > qRow) sv = -3.0e38f;
          s[kt].f[v] = sv;
        }

      // Online softmax (row reductions across 16 lanes of each half-wave).
      unsigned short* Pw = Ps + wave * 16 * PST;
#pragma unroll
      for (int v = 0; v < 8; ++v) {
        float rm = fmaxf(s[0].f[v], s[1].f[v]);
#pragma unroll
        for (int off = 1; off < 16; off <<= 1)
          rm = fmaxf(rm, __shfl_xor(rm, off, 32));
        float mnew = fmaxf(mrow[v], rm);
        float alpha = __expf(mrow[v] - mnew);
        float p0 = __expf(s[0].f[v] - mnew);
        float p1 = __expf(s[1].f[v] - mnew);
        float rs = p0 + p1;
#pragma unroll
        for (int off = 1; off < 16; off <<= 1)
          rs += __shfl_xor(rs, off, 32);
        lrow[v] = lrow[v] * alpha + rs;
        mrow[v] = mnew;
#pragma unroll
        for (int nj = 0; nj < 3; ++nj) o[nj].f[v] *= alpha;
        int m = hf * 8 + v;
        Pw[m * PST + r] = f2bf(p0);             // C-layout -> A-layout via LDS
        Pw[m * PST + 16 + r] = f2bf(p1);
      }

      // O += P x V   (P: 16x32 A-frag, V: 32x48 as three B-frags from Vt)
      v16bf pa = ldsFragA(Pw, PST);
#pragma unroll
      for (int nj = 0; nj < 3; ++nj) {
        v16bf vbf = ldsFragB(Vcur + nj * 16 * VST, VST);
        o[nj].v = __builtin_amdgcn_wmma_f32_16x16x32_bf16(false, pa, false, vbf, (short)0, o[nj].v, false, false);
      }
    }

    // Complete next chunk's V transpose into LDS (loads landed during compute).
    if (haveNext) {
#pragma unroll
      for (int i = 0; i < 3; ++i) {
        int dw = tid + i * 256;
        int key = dw / 24, dp = dw % 24;
        Vnxt[(dp * 2 + 0) * VST + key] = (unsigned short)(vb[i] & 0xFFFFu);
        Vnxt[(dp * 2 + 1) * VST + key] = (unsigned short)(vb[i] >> 16);
      }
    }
  }

  // Normalize and store y[b,t, h*48+d] (bf16).
#pragma unroll
  for (int v = 0; v < 8; ++v) {
    float inv = 1.0f / lrow[v];
    size_t rowOff = (size_t)(b * 2048 + q0 + hf * 8 + v) * 672;
#pragma unroll
    for (int nj = 0; nj < 3; ++nj)
      y[rowOff + h * 48 + nj * 16 + r] = f2bf(o[nj].f[v] * inv);
  }
}

// ---------------------------------------------------------------------------
// Launch
// ---------------------------------------------------------------------------
extern "C" void kernel_launch(void* const* d_in, const int* in_sizes, int n_in,
                              void* d_out, int out_size, void* d_ws, size_t ws_size,
                              hipStream_t stream) {
  (void)in_sizes; (void)n_in; (void)out_size; (void)ws_size;
  constexpr int Bb = 4, Tt = 2048, Cc = 672;
  constexpr int Mm = Bb * Tt;          // 8192
  constexpr int N1 = 3 * Cc;           // 2016

  const float* x      = (const float*)d_in[0];
  const float* w_attn = (const float*)d_in[1];
  const float* w_proj = (const float*)d_in[2];
  unsigned short* out = (unsigned short*)d_out;
  unsigned short* ws  = (unsigned short*)d_ws;

  // Workspace layout (bf16 elements)
  const size_t OFF_XB    = 0;                              // [8192 x 672]
  const size_t OFF_WQKV  = OFF_XB   + (size_t)Mm * Cc;     // [2016 x 672] (N-major)
  const size_t OFF_WPROJ = OFF_WQKV + (size_t)N1 * Cc;     // [672 x 672]  (N-major)
  const size_t OFF_QKV   = OFF_WPROJ + (size_t)Cc * Cc;    // [8192 x 2016]
  const size_t OFF_Y     = OFF_QKV  + (size_t)Mm * N1;     // [8192 x 672]

  unsigned short* xb    = ws + OFF_XB;
  unsigned short* wqkvT = ws + OFF_WQKV;
  unsigned short* wprjT = ws + OFF_WPROJ;
  unsigned short* qkv   = ws + OFF_QKV;
  unsigned short* yb    = ws + OFF_Y;

  cvt_f32_bf16<<<1024, 256, 0, stream>>>(x, xb, Mm * Cc);
  cvt_transpose_bf16<<<1024, 256, 0, stream>>>(w_attn, wqkvT, Cc, N1);
  cvt_transpose_bf16<<<512, 256, 0, stream>>>(w_proj, wprjT, Cc, Cc);

  gemm_bf16_wmma<<<dim3(N1 / 32, Mm / 512), 256, 0, stream>>>(xb, wqkvT, qkv, Cc, N1);
  attn_wmma<<<dim3(Tt / 128, 14, Bb), 256, 0, stream>>>(qkv, yb);
  gemm_bf16_wmma<<<dim3(Cc / 32, Mm / 512), 256, 0, stream>>>(yb, wprjT, out, Cc, Cc);
}
